// StyleDecoder2D_52149492908775
// MI455X (gfx1250) — compile-verified, matrix operand order checked by reading, not run on hardware
//
#include <hip/hip_runtime.h>

// ---------------------------------------------------------------------------
// StyleGAN2-modulated MLP decoder, fully fused for gfx1250 (MI455X).
// Prep kernels swizzle fp32 weights -> bf16 WMMA B-fragment layout in d_ws
// (scales folded in). Main kernel: one 256-thread WG (8 wave32) pushes 32
// samples through all 10 layers with v_wmma_f32_16x16x32_bf16; activations
// and per-sample modulation gammas live in LDS as bf16.
// f32->bf16 conversions use v_fma_mixlo/mixhi_bf16 (1 VALU op each).
// ---------------------------------------------------------------------------

typedef __attribute__((ext_vector_type(16))) __bf16 v16bf;
typedef __attribute__((ext_vector_type(8)))  float  v8f;

union FragU { v16bf v; unsigned int u[8]; };

#define ZDIM 128
#define ROWS 32           // samples per workgroup
#define ZS   136          // LDS stride for z tile   (136*2B = 17*16B aligned)
#define XS   328          // LDS stride for x/gamma  (328*2B = 41*16B aligned)

// f32 -> bf16 via mixed-precision FMA (converts with RNE, writes 16-bit half)
__device__ __forceinline__ unsigned short bfcvt1(float f) {
  unsigned int d;
  asm("v_fma_mixlo_bf16 %0, %1, 1.0, 0" : "=v"(d) : "v"(f));
  return (unsigned short)d;
}
__device__ __forceinline__ unsigned int bfcvt_pair(float lo, float hi) {
  unsigned int d;
  asm("v_fma_mixlo_bf16 %0, %1, 1.0, 0\n\t"
      "v_fma_mixhi_bf16 %0, %2, 1.0, 0"
      : "=v"(d)
      : "v"(lo), "v"(hi));
  return d;
}

// packed bf16 multiply (VOP3P) -- one instruction for two bf16 products
__device__ __forceinline__ unsigned int pk_mul_bf16(unsigned int a, unsigned int b) {
  unsigned int d;
  asm("v_pk_mul_bf16 %0, %1, %2" : "=v"(d) : "v"(a), "v"(b));
  return d;
}

__device__ __forceinline__ v8f wmma_bf16(v16bf a, v16bf b, v8f c) {
  return __builtin_amdgcn_wmma_f32_16x16x32_bf16(false, a, false, b, (short)0, c,
                                                 false, false);
}

// --- A fragment (16x32 bf16) from LDS via 2x ds_load_b128 -------------------
// ISA 7.12.2: lane m<16: K 0..7,16..23 ; lane m+16: K 8..15,24..31 (pairs/dword)
__device__ __forceinline__ v16bf load_a_plain(const unsigned short* p0, int S,
                                              int kb, int lane) {
  const int hf = (lane >> 4) & 1, m = lane & 15;
  const unsigned short* p = p0 + m * S + kb + hf * 8;
  const uint4 lo = *(const uint4*)(p);
  const uint4 hi = *(const uint4*)(p + 16);
  FragU f;
  f.u[0] = lo.x; f.u[1] = lo.y; f.u[2] = lo.z; f.u[3] = lo.w;
  f.u[4] = hi.x; f.u[5] = hi.y; f.u[6] = hi.z; f.u[7] = hi.w;
  return f.v;
}

// --- A fragments of (x*gamma) and gamma^2, fused via v_pk_mul_bf16 ----------
template <bool SQ>
__device__ __forceinline__ void load_a_modsq(const unsigned short* x0,
                                             const unsigned short* g0, int S,
                                             int kb, int lane, v16bf& axm,
                                             v16bf& ag2) {
  const int hf = (lane >> 4) & 1, m = lane & 15;
  const unsigned short* px = x0 + m * S + kb + hf * 8;
  const unsigned short* pg = g0 + m * S + kb + hf * 8;
  const uint4 xlo = *(const uint4*)(px), xhi = *(const uint4*)(px + 16);
  const uint4 glo = *(const uint4*)(pg), ghi = *(const uint4*)(pg + 16);
  const unsigned int xu[8] = {xlo.x, xlo.y, xlo.z, xlo.w, xhi.x, xhi.y, xhi.z, xhi.w};
  const unsigned int gu[8] = {glo.x, glo.y, glo.z, glo.w, ghi.x, ghi.y, ghi.z, ghi.w};
  FragU fm, fs;
#pragma unroll
  for (int r = 0; r < 8; ++r) {
    fm.u[r] = pk_mul_bf16(xu[r], gu[r]);
    if (SQ) fs.u[r] = pk_mul_bf16(gu[r], gu[r]);
  }
  axm = fm.v;
  if (SQ) ag2 = fs.v;
}

// --- B fragment from pre-swizzled bf16 weights: 2x global_load_b128 ---------
// block = 16(N) x 32(K) tile = 256 dwords; word (lane*8 + r) holds the pair
// this lane's fragment register r needs.
__device__ __forceinline__ v16bf load_b_swz(const unsigned int* __restrict__ base,
                                            int blk, int lane) {
  const unsigned int* p = base + blk * 256 + lane * 8;
  const uint4 lo = *(const uint4*)(p);
  const uint4 hi = *(const uint4*)(p + 4);
  FragU f;
  f.u[0] = lo.x; f.u[1] = lo.y; f.u[2] = lo.z; f.u[3] = lo.w;
  f.u[4] = hi.x; f.u[5] = hi.y; f.u[6] = hi.z; f.u[7] = hi.w;
  return f.v;
}

// ---------------------------------------------------------------------------
// Prep: fp32 weights [cols][ld] row-major -> bf16 fragment-swizzled blocks.
// Indices are CLAMPED (not zeroed): padded-K rows multiply a zero A column
// (gamma forced to 0 there), padded-N columns are never read back.
// ---------------------------------------------------------------------------
__global__ void prep_swz(const float* __restrict__ src,
                         unsigned int* __restrict__ dst,
                         unsigned int* __restrict__ dsq, int ld, int klim,
                         int colmax, int ktiles, int total, float scale) {
  const int idx = blockIdx.x * 256 + threadIdx.x;
  if (idx >= total) return;
  const int r = idx & 7;
  const int lane = (idx >> 3) & 31;
  const int blk = idx >> 8;
  const int ktile = blk % ktiles;
  const int ntile = blk / ktiles;
  const int hf = (lane >> 4) & 1, n = lane & 15;
  int col = ntile * 16 + n;
  if (col > colmax - 1) col = colmax - 1;
  int k = ktile * 32 + ((r & 4) ? 16 : 0) + hf * 8 + (r & 3) * 2;
  if (k > klim - 2) k = klim - 2;
  const float* p = src + (long)col * ld + k;
  const float a = p[0] * scale, b = p[1] * scale;
  dst[idx] = bfcvt_pair(a, b);
  if (dsq) dsq[idx] = bfcvt_pair(a * a, b * b);
}

// ---------------------------------------------------------------------------
template <int IN_REAL, int IN_PAD, int OUT_REAL, int OUT_PAD, bool DEMOD,
          bool ACT, bool LAST>
__device__ __forceinline__ void run_layer(
    const unsigned int* __restrict__ Wz,   // bf16 swz:  W * w_scale
    const unsigned int* __restrict__ Wq,   // bf16 swz: (W * w_scale)^2
    const unsigned int* __restrict__ Mz,   // bf16 swz: mw * z_scale
    const float* __restrict__ MB, const float* __restrict__ BS,
    unsigned short* xb, unsigned short* gb, const unsigned short* zl,
    float* __restrict__ out, int row0, int tid) {
  const int lane = tid & 31;
  const int wave = tid >> 5;
  const int hf = (lane >> 4) & 1;
  const int nlane = lane & 15;

  // ------------- gamma = z @ (mw^T * zscale) + mb  ->  LDS bf16 ------------
  constexpr int NT_IN = IN_PAD / 16;
  for (int t = wave; t < 2 * NT_IN; t += 8) {   // 2*NT_IN in {8,32,40}
    const int rg = t & 1;
    const int nt = t >> 1;
    const int col = nt * 16 + nlane;
    const float mbv = MB[col < IN_REAL ? col : IN_REAL - 1];
    v8f acc;
#pragma unroll
    for (int i = 0; i < 8; ++i) acc[i] = mbv;   // fold +mb into accumulator
#pragma unroll
    for (int kk = 0; kk < ZDIM / 32; ++kk) {
      const v16bf a = load_a_plain(zl + rg * 16 * ZS, ZS, kk * 32, lane);
      const v16bf b = load_b_swz(Mz, nt * 4 + kk, lane);
      acc = wmma_bf16(a, b, acc);
    }
#pragma unroll
    for (int r = 0; r < 8; ++r) {
      const int m = rg * 16 + hf * 8 + r;
      const float g = (col < IN_REAL) ? acc[r] : 0.f;  // zero padded gammas
      gb[m * XS + col] = bfcvt1(g);
    }
  }
  __syncthreads();

  // ---- y = (x*gamma)@(W ws)^T ; dacc = (gamma^2)@((W ws)^2)^T (scales folded)
  const int rg = wave & 1;                      // 2 row-groups x 4 col-groups
  const int cg = wave >> 1;
  const int colb0 = cg * 64;
  constexpr int KT = IN_PAD / 32;
  constexpr int NT = OUT_PAD / 16;              // 16 (1 for last layer)
  v8f accY[NT], accD[NT];
#pragma unroll
  for (int nt = 0; nt < NT; ++nt)
#pragma unroll
    for (int i = 0; i < 8; ++i) { accY[nt][i] = 0.f; accD[nt][i] = 0.f; }

  if (colb0 < OUT_PAD) {
#pragma unroll
    for (int kk = 0; kk < KT; ++kk) {
      v16bf axm, ag2;
      load_a_modsq<DEMOD>(xb + rg * 16 * XS, gb + rg * 16 * XS, XS, kk * 32,
                          lane, axm, ag2);
#pragma unroll
      for (int nt = 0; nt < NT; ++nt) {
        const int blk = (cg * 4 + nt) * KT + kk;
        accY[nt] = wmma_bf16(axm, load_b_swz(Wz, blk, lane), accY[nt]);
        if constexpr (DEMOD)
          accD[nt] = wmma_bf16(ag2, load_b_swz(Wq, blk, lane), accD[nt]);
      }
    }
  }
  __syncthreads();  // all waves finished READING xb/gb before xb is rewritten

  if (colb0 < OUT_PAD) {
#pragma unroll
    for (int nt = 0; nt < NT; ++nt) {
      const int col = colb0 + nt * 16 + nlane;
      if (col < OUT_REAL) {
        const float bias = BS[col];
#pragma unroll
        for (int r = 0; r < 8; ++r) {
          const int m = rg * 16 + hf * 8 + r;
          float y = accY[nt][r];
          if constexpr (DEMOD) y *= rsqrtf(accD[nt][r] + 1e-8f);
          y += bias;
          if constexpr (ACT)
            y = fmaxf(y, 0.2f * y) * 1.4142135623730951f;  // FusedLeakyReLU
          if constexpr (LAST)
            out[(long)(row0 + m) * 3 + col] = 1.f / (1.f + __expf(-y));
          else
            xb[m * XS + col] = bfcvt1(y);
        }
      }
    }
  }
  __syncthreads();  // xb visible before next layer's modulation reads
}

// ---------------------------------------------------------------------------
struct Params {
  const float* z;
  const float* pts;
  const unsigned int* Wz[10];
  const unsigned int* Wq[10];
  const unsigned int* Mz[10];
  const float* MB[10];
  const float* BS[10];
  float* out;
};

__global__ void __launch_bounds__(256)
style_decoder_kernel(Params p) {
  __shared__ __align__(16) unsigned short zl[ROWS * ZS];
  __shared__ __align__(16) unsigned short xb[ROWS * XS];
  __shared__ __align__(16) unsigned short gb[ROWS * XS];

  const int tid = threadIdx.x;
  const int row0 = blockIdx.x * ROWS;

  // stage z tile to LDS as bf16, two elements per iteration
  for (int i = tid; i < ROWS * (ZDIM / 2); i += 256) {
    const int r = i >> 6, c2 = (i & 63) * 2;
    const float2 zz = *(const float2*)&p.z[(long)(row0 + r) * ZDIM + c2];
    *(unsigned int*)&zl[r * ZS + c2] = bfcvt_pair(zz.x, zz.y);
  }

  // posenc: 42 real cols zero-padded to 64; copy at [0,64) for layer 0 and at
  // [256,320) for the layer-4 skip (cols >=256 untouched by 256-wide layers).
  if (tid < ROWS) {
    const int r = tid;
    const float x0 = p.pts[(long)(row0 + r) * 2 + 0];
    const float x1 = p.pts[(long)(row0 + r) * 2 + 1];
    unsigned short e[64];
    e[0] = bfcvt1(x0);
    e[1] = bfcvt1(x1);
    float fr = 3.14159265358979323f;
#pragma unroll
    for (int i = 0; i < 10; ++i) {
      e[2 + 4 * i + 0] = bfcvt1(__sinf(fr * x0));
      e[2 + 4 * i + 1] = bfcvt1(__sinf(fr * x1));
      e[2 + 4 * i + 2] = bfcvt1(__cosf(fr * x0));
      e[2 + 4 * i + 3] = bfcvt1(__cosf(fr * x1));
      fr *= 2.f;
    }
#pragma unroll
    for (int i = 42; i < 64; ++i) e[i] = 0;
#pragma unroll
    for (int i = 0; i < 64; ++i) {
      xb[r * XS + i] = e[i];
      xb[r * XS + 256 + i] = e[i];
    }
  }
  __syncthreads();

  run_layer< 42,  64, 256, 256, true,  true,  false>(p.Wz[0], p.Wq[0], p.Mz[0], p.MB[0], p.BS[0], xb, gb, zl, p.out, row0, tid);
  run_layer<256, 256, 256, 256, true,  true,  false>(p.Wz[1], p.Wq[1], p.Mz[1], p.MB[1], p.BS[1], xb, gb, zl, p.out, row0, tid);
  run_layer<256, 256, 256, 256, true,  true,  false>(p.Wz[2], p.Wq[2], p.Mz[2], p.MB[2], p.BS[2], xb, gb, zl, p.out, row0, tid);
  run_layer<256, 256, 256, 256, true,  true,  false>(p.Wz[3], p.Wq[3], p.Mz[3], p.MB[3], p.BS[3], xb, gb, zl, p.out, row0, tid);
  run_layer<298, 320, 256, 256, true,  true,  false>(p.Wz[4], p.Wq[4], p.Mz[4], p.MB[4], p.BS[4], xb, gb, zl, p.out, row0, tid);
  run_layer<256, 256, 256, 256, true,  true,  false>(p.Wz[5], p.Wq[5], p.Mz[5], p.MB[5], p.BS[5], xb, gb, zl, p.out, row0, tid);
  run_layer<256, 256, 256, 256, true,  true,  false>(p.Wz[6], p.Wq[6], p.Mz[6], p.MB[6], p.BS[6], xb, gb, zl, p.out, row0, tid);
  run_layer<256, 256, 256, 256, true,  true,  false>(p.Wz[7], p.Wq[7], p.Mz[7], p.MB[7], p.BS[7], xb, gb, zl, p.out, row0, tid);
  run_layer<256, 256, 256, 256, true,  true,  false>(p.Wz[8], p.Wq[8], p.Mz[8], p.MB[8], p.BS[8], xb, gb, zl, p.out, row0, tid);
  run_layer<256, 256,   3,  16, false, false, true >(p.Wz[9], p.Wq[9], p.Mz[9], p.MB[9], p.BS[9], xb, gb, zl, p.out, row0, tid);
}

// ---------------------------------------------------------------------------
extern "C" void kernel_launch(void* const* d_in, const int* in_sizes, int n_in,
                              void* d_out, int out_size, void* d_ws,
                              size_t ws_size, hipStream_t stream) {
  (void)n_in; (void)out_size; (void)ws_size;
  static const int INR[10]  = {42, 256, 256, 256, 298, 256, 256, 256, 256, 256};
  static const int INP[10]  = {64, 256, 256, 256, 320, 256, 256, 256, 256, 256};
  static const int OUTR[10] = {256, 256, 256, 256, 256, 256, 256, 256, 256, 3};
  static const int OUTP[10] = {256, 256, 256, 256, 256, 256, 256, 256, 256, 16};
  const float ZSCALE = 0.08838834764831845f;  // 1/sqrt(128)

  unsigned int* ws32 = (unsigned int*)d_ws;
  size_t wzo[10], wqo[10], mzo[10], off = 0;
  for (int L = 0; L < 10; ++L) { wzo[L] = off; off += (size_t)(OUTP[L] / 16) * (INP[L] / 32) * 256; }
  for (int L = 0; L < 10; ++L) { wqo[L] = off; off += (size_t)(OUTP[L] / 16) * (INP[L] / 32) * 256; }
  for (int L = 0; L < 10; ++L) { mzo[L] = off; off += (size_t)(INP[L] / 16) * 4 * 256; }
  // off ~= 717K dwords ~= 2.9 MB of scratch (L2-resident)

  Params p;
  p.z   = (const float*)d_in[0];
  p.pts = (const float*)d_in[1];
  p.out = (float*)d_out;

  for (int L = 0; L < 10; ++L) {
    const float* W  = (const float*)d_in[2 + L];
    const float* MW = (const float*)d_in[12 + L];
    p.MB[L] = (const float*)d_in[22 + L];
    p.BS[L] = (const float*)d_in[32 + L];
    p.Wz[L] = ws32 + wzo[L];
    p.Wq[L] = ws32 + wqo[L];
    p.Mz[L] = ws32 + mzo[L];

    const float wscale = 1.0f / sqrtf((float)INR[L]);
    const int ktW = INP[L] / 32;
    const int totW = (OUTP[L] / 16) * ktW * 256;
    prep_swz<<<(totW + 255) / 256, 256, 0, stream>>>(
        W, ws32 + wzo[L], ws32 + wqo[L], INR[L], INR[L], OUTR[L], ktW, totW,
        wscale);

    const int totM = (INP[L] / 16) * 4 * 256;
    prep_swz<<<(totM + 255) / 256, 256, 0, stream>>>(
        MW, ws32 + mzo[L], nullptr, ZDIM, ZDIM, INR[L], 4, totM, ZSCALE);
  }

  const int B = in_sizes[0] / ZDIM;   // 131072
  dim3 grid(B / ROWS);                // 4096 workgroups
  dim3 block(256);                    // 8 wave32
  style_decoder_kernel<<<grid, block, 0, stream>>>(p);
}
